// AttentionOnlyTransformer_37812892074100
// MI455X (gfx1250) — matrix operand order repaired and checked by reading, not compiled, not run on hardware
//
#include <hip/hip_runtime.h>

// ---------------- model constants ----------------
#define BATCH   2
#define SEQ     2048
#define DMODEL  1024
#define NHEADS  16
#define DH      64
#define VOCAB   50257
#define NLAYERS 4
#define MROWS   (BATCH * SEQ)   // 4096

typedef __attribute__((ext_vector_type(16))) __bf16 v16bf;
typedef __attribute__((ext_vector_type(8)))  __bf16 v8bf;
typedef __attribute__((ext_vector_type(8)))  float  v8f;

// A-fragment (16x32 bf16, row-major source): lane holds row m=L&15,
// K [16h,16h+16), h=L>>4.  Two 16B loads.
static __device__ __forceinline__ v16bf load_a_frag(const __bf16* rowbase, int h2) {
    v16bf a;
    ((v8bf*)&a)[0] = *(const v8bf*)(rowbase + 16 * h2);
    ((v8bf*)&a)[1] = *(const v8bf*)(rowbase + 16 * h2 + 8);
    return a;
}
// B-fragment (32x16 bf16) from Bt[N][K] (i.e. column n is contiguous over K):
// lane holds col n=L&15, K [16h,16h+16).  Two 16B loads.
static __device__ __forceinline__ v16bf load_b_frag(const __bf16* colbase, int h2) {
    v16bf b;
    ((v8bf*)&b)[0] = *(const v8bf*)(colbase + 16 * h2);
    ((v8bf*)&b)[1] = *(const v8bf*)(colbase + 16 * h2 + 8);
    return b;
}

// ---------------- weight transpose + fp32 -> bf16 ----------------
// W layout [L][K][N] (row-major) -> Wt [L][N][K] bf16
__global__ void transpose_convert_kernel(const float* __restrict__ W,
                                         __bf16* __restrict__ Wt,
                                         int K, int N, long long total) {
    long long i = (long long)blockIdx.x * blockDim.x + threadIdx.x;
    if (i >= total) return;
    long long per = (long long)K * N;
    long long l  = i / per;
    long long r  = i % per;
    long long kk = r / N;
    long long nn = r % N;
    Wt[l * per + nn * (long long)K + kk] = (__bf16)W[i];
}

// ---------------- embedding: x = tok_emb[tok] + pe ----------------
__global__ void embed_kernel(const int* __restrict__ tokens,
                             const float* __restrict__ tok_emb,
                             const float* __restrict__ pe,
                             float* __restrict__ xf, __bf16* __restrict__ xb) {
    long long i = (long long)blockIdx.x * blockDim.x + threadIdx.x;
    if (i >= (long long)MROWS * DMODEL) return;
    int d  = (int)(i % DMODEL);
    long long bs = i / DMODEL;
    int s  = (int)(bs % SEQ);
    int t  = tokens[bs];
    float v = tok_emb[(size_t)t * DMODEL + d] + pe[(size_t)s * DMODEL + d];
    xf[i] = v;
    xb[i] = (__bf16)v;
}

// ---------------- generic bf16 WMMA GEMM ----------------
// C[M][N] = A[M][K](bf16,row-major) x Bt[N][K](bf16)  (+ resid), one wave per
// 64x32 tile: 4 A-frags x 2 B-frags -> 8 wmma per 32-wide K step.
__global__ void gemm_bf16_kernel(const __bf16* __restrict__ A,
                                 const __bf16* __restrict__ Bt,
                                 float* __restrict__ Cf,          // fp32 out (or null)
                                 __bf16* __restrict__ Cb,         // bf16 out (or null)
                                 const float* __restrict__ resid, // fp32 add (or null)
                                 int M, int N, int K) {
    const int lane = threadIdx.x & 31;
    const int nl = lane & 15, h2 = lane >> 4;
    const int n0 = blockIdx.x * 32;
    const int m0 = blockIdx.y * 64;

    // loop-invariant row bases (keeps addressing in SGPR+offset form, less
    // per-iteration VALU address math / register pressure)
    const __bf16* brow[2];
#pragma unroll
    for (int j = 0; j < 2; ++j) {
        int n = n0 + j * 16 + nl;
        if (n >= N) n = N - 1;                    // clamp (store is guarded)
        brow[j] = Bt + (size_t)n * K;
    }
    const __bf16* arow[4];
#pragma unroll
    for (int i = 0; i < 4; ++i)
        arow[i] = A + (size_t)(m0 + i * 16 + nl) * K;

    v8f acc[4][2];
#pragma unroll
    for (int i = 0; i < 4; ++i)
#pragma unroll
        for (int j = 0; j < 2; ++j)
#pragma unroll
            for (int r = 0; r < 8; ++r) acc[i][j][r] = 0.f;

    for (int k0 = 0; k0 < K; k0 += 32) {
        // prefetch the streaming operand (weights) ~4 K-steps ahead;
        // lowers to global_prefetch_b8 (speculative: OOB is silently dropped)
#pragma unroll
        for (int j = 0; j < 2; ++j)
            __builtin_prefetch(brow[j] + k0 + 128, 0, 3);

        v16bf bfrag[2];
#pragma unroll
        for (int j = 0; j < 2; ++j)
            bfrag[j] = load_b_frag(brow[j] + k0, h2);
        v16bf afrag[4];
#pragma unroll
        for (int i = 0; i < 4; ++i)
            afrag[i] = load_a_frag(arow[i] + k0, h2);
#pragma unroll
        for (int i = 0; i < 4; ++i)
#pragma unroll
            for (int j = 0; j < 2; ++j)
                acc[i][j] = __builtin_amdgcn_wmma_f32_16x16x32_bf16(
                    false, afrag[i], false, bfrag[j], (short)0, acc[i][j],
                    false, false);
    }

#pragma unroll
    for (int i = 0; i < 4; ++i)
#pragma unroll
        for (int j = 0; j < 2; ++j) {
            int n = n0 + j * 16 + nl;
            if (n >= N) continue;
#pragma unroll
            for (int r = 0; r < 8; ++r) {
                int m = m0 + i * 16 + r + 8 * h2;
                float v = acc[i][j][r];
                size_t idx = (size_t)m * N + n;
                if (resid) v += resid[idx];
                if (Cf) Cf[idx] = v;
                if (Cb) Cb[idx] = (__bf16)v;
            }
        }
}

// ---------------- qkv reshape: [B,S,3D] -> q/k [BH,S,64], vT [BH,64,S] ----------------
__global__ void reshape_qkv_kernel(const __bf16* __restrict__ qkv,
                                   __bf16* __restrict__ q,
                                   __bf16* __restrict__ k,
                                   __bf16* __restrict__ vT) {
    long long i = (long long)blockIdx.x * blockDim.x + threadIdx.x;
    if (i >= (long long)MROWS * DMODEL) return;
    int d  = (int)(i % DMODEL);
    long long bs = i / DMODEL;
    int s  = (int)(bs % SEQ);
    int b  = (int)(bs / SEQ);
    int hh = d / DH, dd = d % DH;
    size_t src = (size_t)bs * (3 * DMODEL);
    size_t bh  = (size_t)b * NHEADS + hh;
    q[(bh * SEQ + s) * DH + dd]  = qkv[src + d];
    k[(bh * SEQ + s) * DH + dd]  = qkv[src + DMODEL + d];
    vT[(bh * DH + dd) * SEQ + s] = qkv[src + 2 * DMODEL + d];
}

// ---------------- flash-style causal attention ----------------
// One wave per (b,h,16-query-row tile).  Scores and P*V via WMMA bf16;
// online softmax state + P staged in LDS.
__global__ void attn_kernel(const __bf16* __restrict__ q,
                            const __bf16* __restrict__ k,
                            const __bf16* __restrict__ vT,
                            __bf16* __restrict__ out /* [B,S,D] bf16 */) {
    __shared__ __attribute__((aligned(32))) float  sc[16][32];
    __shared__ __attribute__((aligned(32))) __bf16 pl[16][32];
    __shared__ float mi[16], li[16], al[16];

    const int lane = threadIdx.x & 31;
    const int nl = lane & 15, h2 = lane >> 4;
    int bid = blockIdx.x;
    const int qt = bid % (SEQ / 16); bid /= (SEQ / 16);
    const int h  = bid % NHEADS;     bid /= NHEADS;
    const int b  = bid;
    const int q0 = qt * 16;

    const size_t bh = (size_t)b * NHEADS + h;
    const __bf16* qbase = q  + bh * SEQ * DH;
    const __bf16* kbase = k  + bh * SEQ * DH;
    const __bf16* vbase = vT + bh * DH * SEQ;

    v16bf qa[2];
#pragma unroll
    for (int c = 0; c < 2; ++c)
        qa[c] = load_a_frag(qbase + (size_t)(q0 + nl) * DH + 32 * c, h2);

    v8f o[4];
#pragma unroll
    for (int t = 0; t < 4; ++t)
#pragma unroll
        for (int r = 0; r < 8; ++r) o[t][r] = 0.f;

    if (lane < 16) { mi[lane] = -1e30f; li[lane] = 0.f; }
    __syncthreads();

    const int kmax = q0 + 16;
    const int nb   = (kmax + 31) / 32;
    for (int jb = 0; jb < nb; ++jb) {
        // ---- scores: two 16x16 tiles covering 32 keys ----
#pragma unroll
        for (int sub = 0; sub < 2; ++sub) {
            int kb  = jb * 32 + sub * 16;
            int key = kb + nl;
            int keyc = key < SEQ ? key : SEQ - 1;
            v8f s;
#pragma unroll
            for (int r = 0; r < 8; ++r) s[r] = 0.f;
#pragma unroll
            for (int c = 0; c < 2; ++c) {
                v16bf kf = load_b_frag(kbase + (size_t)keyc * DH + 32 * c, h2);
                s = __builtin_amdgcn_wmma_f32_16x16x32_bf16(
                        false, qa[c], false, kf, (short)0, s, false, false);
            }
#pragma unroll
            for (int r = 0; r < 8; ++r) {
                int m = r + 8 * h2;
                float sv = s[r] * 0.125f;                 // 1/sqrt(64)
                if (key > q0 + m) sv = -1e30f;            // causal mask
                sc[m][sub * 16 + nl] = sv;
            }
        }
        __syncthreads();

        // ---- online softmax row pass (16 lanes, one row each) ----
        if (lane < 16) {
            int m = lane;
            float mo = mi[m], mx = mo;
            for (int t = 0; t < 32; ++t) { float s = sc[m][t]; mx = s > mx ? s : mx; }
            float a = __expf(mo - mx);
            float sum = 0.f;
            for (int t = 0; t < 32; ++t) {
                float p = __expf(sc[m][t] - mx);
                pl[m][t] = (__bf16)p;
                sum += p;
            }
            li[m] = li[m] * a + sum;
            mi[m] = mx;
            al[m] = a;
        }
        __syncthreads();

        // ---- rescale running output, then O += P x V ----
#pragma unroll
        for (int t = 0; t < 4; ++t)
#pragma unroll
            for (int r = 0; r < 8; ++r) o[t][r] *= al[r + 8 * h2];

        v16bf pf = load_a_frag(&pl[nl][0], h2);           // P as A-frag (K=32 keys)
        int kv = jb * 32 + 16 * h2;
        if (kv > SEQ - 16) kv = SEQ - 16;                 // belt & braces
#pragma unroll
        for (int t = 0; t < 4; ++t) {
            int dc = t * 16 + nl;
            v16bf vf;
            ((v8bf*)&vf)[0] = *(const v8bf*)(vbase + (size_t)dc * SEQ + kv);
            ((v8bf*)&vf)[1] = *(const v8bf*)(vbase + (size_t)dc * SEQ + kv + 8);
            o[t] = __builtin_amdgcn_wmma_f32_16x16x32_bf16(
                       false, pf, false, vf, (short)0, o[t], false, false);
        }
        __syncthreads();
    }

    // ---- normalize and store (bf16, back to [B,S,H*dh]) ----
#pragma unroll
    for (int t = 0; t < 4; ++t)
#pragma unroll
        for (int r = 0; r < 8; ++r) {
            int m = r + 8 * h2;
            float v = o[t][r] / li[m];
            size_t row = (size_t)b * SEQ + q0 + m;
            out[row * DMODEL + h * DH + t * 16 + nl] = (__bf16)v;
        }
}

// ---------------- host orchestration ----------------
extern "C" void kernel_launch(void* const* d_in, const int* in_sizes, int n_in,
                              void* d_out, int out_size, void* d_ws, size_t ws_size,
                              hipStream_t stream) {
    const int*   tokens  = (const int*)d_in[0];
    const float* tok_emb = (const float*)d_in[1];
    const float* W_qkv   = (const float*)d_in[2];
    const float* W_o     = (const float*)d_in[3];
    const float* W_un    = (const float*)d_in[4];
    const float* pe      = (const float*)d_in[5];
    float* out = (float*)d_out;
    (void)in_sizes; (void)n_in; (void)out_size; (void)ws_size;

    char* ws = (char*)d_ws;
    auto alloc = [&](size_t bytes) {
        char* p = ws;
        ws += (bytes + 255) & ~(size_t)255;
        return p;
    };
    float*  xf    = (float*)alloc((size_t)MROWS * DMODEL * 4);
    __bf16* xb    = (__bf16*)alloc((size_t)MROWS * DMODEL * 2);
    __bf16* qkv   = (__bf16*)alloc((size_t)MROWS * 3 * DMODEL * 2);
    __bf16* qb    = (__bf16*)alloc((size_t)MROWS * DMODEL * 2);
    __bf16* kb    = (__bf16*)alloc((size_t)MROWS * DMODEL * 2);
    __bf16* vT    = (__bf16*)alloc((size_t)MROWS * DMODEL * 2);
    __bf16* ab    = (__bf16*)alloc((size_t)MROWS * DMODEL * 2);
    __bf16* wqkvt = (__bf16*)alloc((size_t)NLAYERS * 3 * DMODEL * DMODEL * 2);
    __bf16* wot   = (__bf16*)alloc((size_t)NLAYERS * DMODEL * DMODEL * 2);
    __bf16* wunt  = (__bf16*)alloc((size_t)VOCAB * DMODEL * 2);

    // 1) weights -> bf16 transposed [N][K]
    {
        long long tot = (long long)NLAYERS * DMODEL * (3 * DMODEL);
        transpose_convert_kernel<<<(unsigned)((tot + 255) / 256), 256, 0, stream>>>(
            W_qkv, wqkvt, DMODEL, 3 * DMODEL, tot);
    }
    {
        long long tot = (long long)NLAYERS * DMODEL * DMODEL;
        transpose_convert_kernel<<<(unsigned)((tot + 255) / 256), 256, 0, stream>>>(
            W_o, wot, DMODEL, DMODEL, tot);
    }
    {
        long long tot = (long long)DMODEL * VOCAB;
        transpose_convert_kernel<<<(unsigned)((tot + 255) / 256), 256, 0, stream>>>(
            W_un, wunt, DMODEL, VOCAB, tot);
    }

    // 2) embedding
    {
        long long tot = (long long)MROWS * DMODEL;
        embed_kernel<<<(unsigned)((tot + 255) / 256), 256, 0, stream>>>(
            tokens, tok_emb, pe, xf, xb);
    }

    // 3) transformer layers
    for (int l = 0; l < NLAYERS; ++l) {
        dim3 g1((3 * DMODEL) / 32, MROWS / 64);
        gemm_bf16_kernel<<<g1, 32, 0, stream>>>(
            xb, wqkvt + (size_t)l * 3 * DMODEL * DMODEL,
            nullptr, qkv, nullptr, MROWS, 3 * DMODEL, DMODEL);

        long long tot = (long long)MROWS * DMODEL;
        reshape_qkv_kernel<<<(unsigned)((tot + 255) / 256), 256, 0, stream>>>(
            qkv, qb, kb, vT);

        attn_kernel<<<BATCH * NHEADS * (SEQ / 16), 32, 0, stream>>>(qb, kb, vT, ab);

        dim3 g2(DMODEL / 32, MROWS / 64);
        gemm_bf16_kernel<<<g2, 32, 0, stream>>>(
            ab, wot + (size_t)l * DMODEL * DMODEL,
            xf, xb, xf, MROWS, DMODEL, DMODEL);   // residual add, refresh bf16 x
    }

    // 4) unembed -> logits (fp32, store-bound: 824 MB)
    dim3 g3((VOCAB + 31) / 32, MROWS / 64);
    gemm_bf16_kernel<<<g3, 32, 0, stream>>>(
        xb, wunt, out, nullptr, nullptr, MROWS, VOCAB, DMODEL);
}
